// TwoPathSigmaLowRankLinear_2259152798356
// MI455X (gfx1250) — 8-bit matrix operand order repaired and checked by reading, compiled, not run on hardware
//
#include <hip/hip_runtime.h>
#include <stdint.h>

#define NTOK  8192
#define IN_F  4096
#define OUT_F 4096
#define RANK  1024

typedef __attribute__((ext_vector_type(8))) int v8i;
typedef int gvec4 __attribute__((vector_size(16)));

#define GLOBAL_AS __attribute__((address_space(1)))
#define LDS_AS    __attribute__((address_space(3)))

#if defined(__has_builtin)
#  if __has_builtin(__builtin_amdgcn_global_load_async_to_lds_b128)
#    define HAVE_ASYNC_LDS 1
#  endif
#  if __has_builtin(__builtin_amdgcn_s_wait_asynccnt)
#    define WAIT_ASYNC(n) __builtin_amdgcn_s_wait_asynccnt(n)
#  endif
#endif
#ifndef WAIT_ASYNC
#  define WAIT_ASYNC(n) asm volatile("s_wait_asynccnt %0" ::"i"(n) : "memory")
#endif

__global__ void transpose_i8(const int8_t* __restrict__ in,
                             int8_t* __restrict__ out, int K, int N) {
  __shared__ int8_t tile[32][33];
  const int n0 = blockIdx.x * 32;
  const int k0 = blockIdx.y * 32;
  const int tx = threadIdx.x, ty = threadIdx.y;
#pragma unroll
  for (int i = 0; i < 4; ++i)
    tile[ty + 8 * i][tx] = in[(size_t)(k0 + ty + 8 * i) * N + (n0 + tx)];
  __syncthreads();
#pragma unroll
  for (int i = 0; i < 4; ++i)
    out[(size_t)(n0 + ty + 8 * i) * K + (k0 + tx)] = tile[tx][ty + 8 * i];
}

__global__ __launch_bounds__(256) void rowquant(const float* __restrict__ in,
                                                int cols,
                                                int8_t* __restrict__ q,
                                                float* __restrict__ sc) {
  const int row = blockIdx.x;
  const float* r = in + (size_t)row * cols;
  float m = 0.0f;
  for (int c = threadIdx.x; c < cols; c += 256) m = fmaxf(m, fabsf(r[c]));
  __shared__ float red[256];
  red[threadIdx.x] = m;
  __syncthreads();
  for (int s = 128; s > 0; s >>= 1) {
    if (threadIdx.x < s)
      red[threadIdx.x] = fmaxf(red[threadIdx.x], red[threadIdx.x + s]);
    __syncthreads();
  }
  const float absmax = fmaxf(red[0], 1e-8f);
  if (threadIdx.x == 0) sc[row] = absmax / 127.0f;
  const float inv = 127.0f / absmax;
  for (int c = threadIdx.x; c < cols; c += 256) {
    float v = rintf(r[c] * inv);
    v = fminf(fmaxf(v, -127.0f), 127.0f);
    q[(size_t)row * cols + c] = (int8_t)v;
  }
}

__global__ __launch_bounds__(256) void gemm_i8_wmma(
    const int8_t* __restrict__ A, const int8_t* __restrict__ BT,
    const float* __restrict__ rowsc, const float* __restrict__ colsc,
    const float* __restrict__ colsc2, const float* __restrict__ bias,
    float* __restrict__ out, int M, int N, int K, int accumulate) {
  constexpr int LDT = 80;
  __shared__ __align__(16) int8_t lsA[2][128 * LDT];
  __shared__ __align__(16) int8_t lsB[2][128 * LDT];

  const int tid = threadIdx.x;
  const int lane = tid & 31;
  const int wave = tid >> 5;
  const int wm = wave & 1;
  const int wn = wave >> 1;
  const int m0 = wm * 64;
  const int n0 = wn * 32;
  const int half = lane >> 4;
  const int l16 = lane & 15;

  const int mBase = blockIdx.y * 128;
  const int nBase = blockIdx.x * 128;

  const int i0 = tid, i1 = tid + 256;
  const int rA0 = i0 >> 2, cA0 = (i0 & 3) << 4;
  const int rA1 = i1 >> 2, cA1 = (i1 & 3) << 4;

  auto stage = [&](int buf, int k0) {
#if defined(HAVE_ASYNC_LDS)
    __builtin_amdgcn_global_load_async_to_lds_b128(
        (GLOBAL_AS gvec4*)(A + (size_t)(mBase + rA0) * K + k0 + cA0),
        (LDS_AS gvec4*)&lsA[buf][rA0 * LDT + cA0], 0, 0);
    __builtin_amdgcn_global_load_async_to_lds_b128(
        (GLOBAL_AS gvec4*)(BT + (size_t)(nBase + rA0) * K + k0 + cA0),
        (LDS_AS gvec4*)&lsB[buf][rA0 * LDT + cA0], 0, 0);
    __builtin_amdgcn_global_load_async_to_lds_b128(
        (GLOBAL_AS gvec4*)(A + (size_t)(mBase + rA1) * K + k0 + cA1),
        (LDS_AS gvec4*)&lsA[buf][rA1 * LDT + cA1], 0, 0);
    __builtin_amdgcn_global_load_async_to_lds_b128(
        (GLOBAL_AS gvec4*)(BT + (size_t)(nBase + rA1) * K + k0 + cA1),
        (LDS_AS gvec4*)&lsB[buf][rA1 * LDT + cA1], 0, 0);
#else
    *(int4*)&lsA[buf][rA0 * LDT + cA0] =
        *(const int4*)&A[(size_t)(mBase + rA0) * K + k0 + cA0];
    *(int4*)&lsB[buf][rA0 * LDT + cA0] =
        *(const int4*)&BT[(size_t)(nBase + rA0) * K + k0 + cA0];
    *(int4*)&lsA[buf][rA1 * LDT + cA1] =
        *(const int4*)&A[(size_t)(mBase + rA1) * K + k0 + cA1];
    *(int4*)&lsB[buf][rA1 * LDT + cA1] =
        *(const int4*)&BT[(size_t)(nBase + rA1) * K + k0 + cA1];
#endif
  };

  v8i acc[4][2] = {};

  stage(0, 0);
  const int nsteps = K >> 6;
  for (int step = 0; step < nsteps; ++step) {
    const int cur = step & 1;
    const bool more = (step + 1) < nsteps;
    if (more) stage(cur ^ 1, (step + 1) << 6);
#if defined(HAVE_ASYNC_LDS)
    if (more) WAIT_ASYNC(4); else WAIT_ASYNC(0);
#endif
    __syncthreads();

    v8i bfrag[2];
#pragma unroll
    for (int tn = 0; tn < 2; ++tn) {
      const int8_t* br = &lsB[cur][(n0 + tn * 16 + l16) * LDT];
      union { v8i v; int4 q[2]; } b;
      b.q[0] = *(const int4*)(br + half * 16);
      b.q[1] = *(const int4*)(br + 32 + half * 16);
      bfrag[tn] = b.v;
    }

#pragma unroll
    for (int tm = 0; tm < 4; ++tm) {
      const int8_t* ar = &lsA[cur][(m0 + tm * 16 + l16) * LDT];
      union { v8i v; long long l[4]; } a;
      *(int4*)&a.l[0] = *(const int4*)(ar + half * 16);
      *(int4*)&a.l[2] = *(const int4*)(ar + 32 + half * 16);
#pragma unroll
      for (int tn = 0; tn < 2; ++tn) {
        acc[tm][tn] = __builtin_amdgcn_wmma_i32_16x16x64_iu8(
             true, a.v,  true, bfrag[tn], acc[tm][tn],
             false,  false);
      }
    }
    __syncthreads();
  }

#pragma unroll
  for (int tn = 0; tn < 2; ++tn) {
    const int col = nBase + n0 + tn * 16 + l16;
    float cs = colsc[col];
    if (colsc2) cs *= colsc2[col];
    const float bi = bias ? bias[col] : 0.0f;
#pragma unroll
    for (int tm = 0; tm < 4; ++tm) {
#pragma unroll
      for (int r = 0; r < 8; ++r) {
        const int row = mBase + m0 + tm * 16 + (half << 3) + r;
        const float v = (float)acc[tm][tn][r] * rowsc[row] * cs;
        const size_t o = (size_t)row * N + col;
        if (accumulate)
          out[o] += v;
        else
          out[o] = v + bi;
      }
    }
  }
}

extern "C" void kernel_launch(void* const* d_in, const int* in_sizes, int n_in,
                              void* d_out, int out_size, void* d_ws,
                              size_t ws_size, hipStream_t stream) {
  (void)in_sizes; (void)n_in; (void)out_size; (void)ws_size;

  const float*  x      = (const float*)d_in[0];
  const int8_t* vh_q_t = (const int8_t*)d_in[1];
  const float*  vh_s   = (const float*)d_in[2];
  const float*  sh     = (const float*)d_in[3];
  const int8_t* uh_q_t = (const int8_t*)d_in[4];
  const float*  uh_s   = (const float*)d_in[5];
  const int8_t* vl_q_t = (const int8_t*)d_in[6];
  const float*  vl_s   = (const float*)d_in[7];
  const float*  sl     = (const float*)d_in[8];
  const int8_t* ul_q_t = (const int8_t*)d_in[9];
  const float*  ul_s   = (const float*)d_in[10];
  const float*  bias   = (const float*)d_in[11];
  float* out = (float*)d_out;

  char* w = (char*)d_ws;
  int8_t* xq   = (int8_t*)w;                         w += (size_t)NTOK * IN_F;
  float*  xsc  = (float*)w;                          w += (size_t)NTOK * 4;
  float*  t    = (float*)w;                          w += (size_t)NTOK * RANK * 4;
  int8_t* tq   = (int8_t*)w;                         w += (size_t)NTOK * RANK;
  float*  tsc  = (float*)w;                          w += (size_t)NTOK * 4;
  int8_t* WvhT = (int8_t*)w;                         w += (size_t)RANK * IN_F;
  int8_t* WuhT = (int8_t*)w;                         w += (size_t)OUT_F * RANK;
  int8_t* WvlT = (int8_t*)w;                         w += (size_t)RANK * IN_F;
  int8_t* WulT = (int8_t*)w;

  const dim3 tb(32, 8);
  transpose_i8<<<dim3(RANK / 32, IN_F / 32), tb, 0, stream>>>(vh_q_t, WvhT, IN_F, RANK);
  transpose_i8<<<dim3(OUT_F / 32, RANK / 32), tb, 0, stream>>>(uh_q_t, WuhT, RANK, OUT_F);
  transpose_i8<<<dim3(RANK / 32, IN_F / 32), tb, 0, stream>>>(vl_q_t, WvlT, IN_F, RANK);
  transpose_i8<<<dim3(OUT_F / 32, RANK / 32), tb, 0, stream>>>(ul_q_t, WulT, RANK, OUT_F);

  rowquant<<<NTOK, 256, 0, stream>>>(x, IN_F, xq, xsc);

  gemm_i8_wmma<<<dim3(RANK / 128, NTOK / 128), 256, 0, stream>>>(
      xq, WvhT, xsc, vh_s, sh, nullptr, t, NTOK, RANK, IN_F, 0);
  rowquant<<<NTOK, 256, 0, stream>>>(t, RANK, tq, tsc);
  gemm_i8_wmma<<<dim3(OUT_F / 128, NTOK / 128), 256, 0, stream>>>(
      tq, WuhT, tsc, uh_s, nullptr, bias, out, NTOK, OUT_F, RANK, 0);

  gemm_i8_wmma<<<dim3(RANK / 128, NTOK / 128), 256, 0, stream>>>(
      xq, WvlT, xsc, vl_s, sl, nullptr, t, NTOK, RANK, IN_F, 0);
  rowquant<<<NTOK, 256, 0, stream>>>(t, RANK, tq, tsc);
  gemm_i8_wmma<<<dim3(OUT_F / 128, NTOK / 128), 256, 0, stream>>>(
      tq, WulT, tsc, ul_s, nullptr, nullptr, out, NTOK, OUT_F, RANK, 1);
}
